// Seq2Seq_29867202576577
// MI455X (gfx1250) — matrix operand order repaired and checked by reading, not compiled, not run on hardware
//
#include <hip/hip_runtime.h>

// ---------------------------------------------------------------------------
// Seq2Seq LSTM (H=1024, B=32, T=256) persistent-RNN kernel for gfx1250.
//  - bf16 V_WMMA_F32_16X16X32_BF16 for all h@W^T GEMMs
//  - weights pre-packed into WMMA B-fragment order (coalesced 32B/lane loads)
//  - h staged in LDS via GLOBAL_LOAD_ASYNC_TO_LDS_B128 when available
//  - hardware V_TANH_F32 for all gate activations when available
//  - grid-wide sync via atomic counter (32 resident workgroups)
// ---------------------------------------------------------------------------

#define HDIM 1024
#define G4   4096
#define BSZ  32
#define TLEN 256
#define NWG  32
#define TPB  256
#define HP   1032          // LDS pitch for staged h (ushorts), bank-conflict pad
#define GPCH 36            // LDS pitch for gate tiles (floats)
#define MATSZ 4194304      // 4096*1024 elements per packed weight matrix

typedef __attribute__((ext_vector_type(16))) __bf16 v16bf;
typedef __attribute__((ext_vector_type(8)))  float  v8f;
typedef int v4i __attribute__((vector_size(16)));
typedef unsigned short u16;
typedef unsigned int   u32;
typedef unsigned long long u64;

typedef __attribute__((address_space(1))) v4i* gv4p;   // global int4*
typedef __attribute__((address_space(3))) v4i* lv4p;   // LDS int4*

// ---- feature probes --------------------------------------------------------
#if defined(__has_builtin)
#if __has_builtin(__builtin_amdgcn_tanhf)
#define HW_TANH(x) __builtin_amdgcn_tanhf(x)
#elif __has_builtin(__builtin_amdgcn_tanh_f32)
#define HW_TANH(x) __builtin_amdgcn_tanh_f32(x)
#else
#define HW_TANH(x) tanhf(x)
#endif
#if __has_builtin(__builtin_amdgcn_global_load_async_to_lds_b128) && \
    __has_builtin(__builtin_amdgcn_s_wait_asynccnt)
#define HAVE_ASYNC_LDS 1
#endif
#else
#define HW_TANH(x) tanhf(x)
#endif

__device__ __forceinline__ float bf2f(u16 v) {
  union { u32 i; float f; } u; u.i = ((u32)v) << 16; return u.f;
}
__device__ __forceinline__ u16 f2bf(float f) {
  union { float f; u32 i; } u; u.f = f;
  u32 r = u.i + 0x7FFFu + ((u.i >> 16) & 1u);   // round-to-nearest-even
  return (u16)(r >> 16);
}
// sigmoid via hardware tanh: sig(x) = 0.5 + 0.5*tanh(x/2)
__device__ __forceinline__ float sigf(float x) {
  return fmaf(0.5f, HW_TANH(0.5f * x), 0.5f);
}

union V16U { uint4 u[2]; v16bf v; };

// A-fragment (16x32 bf16, M x K): lane L holds row m=L&15; elements 0..7 are
// K = kbase..kbase+7, elements 8..15 are K = kbase+16..kbase+23 (kbase has the
// lane>>4 * 8 offset folded into the pointer).
__device__ __forceinline__ v16bf load_a_frag(const u16* p) {
  V16U t;
  t.u[0] = *(const uint4*)(p);
  t.u[1] = *(const uint4*)(p + 16);
  return t.v;
}

__device__ __forceinline__ v8f wmma_bf16(v16bf a, v16bf b, v8f c) {
  return __builtin_amdgcn_wmma_f32_16x16x32_bf16(false, a, false, b,
                                                 (short)0, c, false, false);
}

// One GEMM pass: acc{0,1} += h(32x1024) @ Wpack_tile(1024x16). 32 k-tiles.
__device__ __forceinline__ void gemm_pass(const u16* __restrict__ wpack,
                                          const u16* __restrict__ hs,
                                          int lane, int nt,
                                          v8f& acc0, v8f& acc1) {
  const int m0 = lane & 15;
  const int kg = (lane >> 4) << 3;         // 0 or 8
  const u16* A0 = hs + m0 * HP + kg;
  const u16* A1 = hs + (m0 + 16) * HP + kg;
  const u16* W  = wpack + (size_t)nt * 16384 + lane * 16;
#pragma unroll 4
  for (int kt = 0; kt < 32; ++kt) {
    v16bf a0 = load_a_frag(A0 + kt * 32);
    v16bf a1 = load_a_frag(A1 + kt * 32);
    v16bf b  = *reinterpret_cast<const v16bf*>(W + kt * 512);
    acc0 = wmma_bf16(a0, b, acc0);
    acc1 = wmma_bf16(a1, b, acc1);
  }
}

// Stage 32x1024 bf16 h matrix from global into padded LDS (64KB).
// Prefers the CDNA5 async direct-to-LDS path (no VGPR round trip, ASYNCcnt).
__device__ __forceinline__ void stage_h(u16* __restrict__ hs,
                                        const u16* __restrict__ hg, int tid) {
  const int r  = tid >> 3;
  const int cb = (tid & 7) << 7;           // *128 ushorts
#ifdef HAVE_ASYNC_LDS
  const u16* src = hg + r * HDIM + cb;
  u16* dst = hs + r * HP + cb;
#pragma unroll
  for (int i = 0; i < 16; ++i) {
    __builtin_amdgcn_global_load_async_to_lds_b128(
        (gv4p)(u64)(src + i * 8),
        (lv4p)(u32)(u64)(dst + i * 8),
        0, 0);
  }
  __builtin_amdgcn_s_wait_asynccnt(0);
#else
  const uint4* src = (const uint4*)(hg + r * HDIM + cb);
  uint4* dst = (uint4*)(hs + r * HP + cb);
#pragma unroll
  for (int i = 0; i < 16; ++i) dst[i] = src[i];
#endif
}

__device__ __forceinline__ void store_gates(float* __restrict__ gates,
                                            const v8f& a0, const v8f& a1,
                                            int gate, int u_l, int mbase) {
  float* gp = gates + (gate * 32 + u_l) * GPCH;
  *(float4*)(gp + mbase)          = make_float4(a0[0], a0[1], a0[2], a0[3]);
  *(float4*)(gp + mbase + 4)      = make_float4(a0[4], a0[5], a0[6], a0[7]);
  *(float4*)(gp + 16 + mbase)     = make_float4(a1[0], a1[1], a1[2], a1[3]);
  *(float4*)(gp + 16 + mbase + 4) = make_float4(a1[4], a1[5], a1[6], a1[7]);
}

// Elementwise LSTM cell update for this WG's 32 hidden units; writes new h
// as one packed 8-byte store (4 bf16) per thread.
__device__ __forceinline__ void cell_update(const float* __restrict__ gates,
                                            float* __restrict__ cS,
                                            u16* __restrict__ hout,
                                            int u0, int tid) {
  const int m  = tid >> 3;
  const int ub = (tid & 7) << 2;
  u16 hv[4];
#pragma unroll
  for (int j = 0; j < 4; ++j) {
    const int u = ub + j;
    const float gi = gates[(0 * 32 + u) * GPCH + m];
    const float gf = gates[(1 * 32 + u) * GPCH + m];
    const float gg = gates[(2 * 32 + u) * GPCH + m];
    const float go = gates[(3 * 32 + u) * GPCH + m];
    const float c  = sigf(gf) * cS[m * 32 + u] + sigf(gi) * HW_TANH(gg);
    cS[m * 32 + u] = c;
    hv[j] = f2bf(sigf(go) * HW_TANH(c));
  }
  const u32 lo = (u32)hv[0] | ((u32)hv[1] << 16);
  const u32 hi = (u32)hv[2] | ((u32)hv[3] << 16);
  *(uint2*)(hout + m * HDIM + u0 + ub) = make_uint2(lo, hi);
}

__device__ __forceinline__ void grid_sync(u32* cnt, u32 target) {
  __threadfence();                          // release prior global stores
  __syncthreads();
  if (threadIdx.x == 0) {
    __hip_atomic_fetch_add(cnt, 1u, __ATOMIC_ACQ_REL, __HIP_MEMORY_SCOPE_AGENT);
    while (__hip_atomic_load(cnt, __ATOMIC_ACQUIRE,
                             __HIP_MEMORY_SCOPE_AGENT) < target) {
      __builtin_amdgcn_s_sleep(2);
    }
  }
  __syncthreads();
  __threadfence();                          // acquire
}

// ---------------------------------------------------------------------------
// Prologue kernels
// ---------------------------------------------------------------------------
__global__ void zero_u32_kernel(u32* p, int n) {
  int i = blockIdx.x * blockDim.x + threadIdx.x;
  if (i < n) p[i] = 0u;
}

// Pack fp32 (4096 x 1024, row-major, used transposed as B) into bf16
// WMMA-B-fragment order: block (nt, kt) holds a 32x16 tile; within a block,
// lane L owns 16 contiguous halfs: n = nt*16 + (L&15),
// k = kt*32 + 16*(i>>3) + 8*(L>>4) + (i&7).
__global__ void pack6_kernel(const float* __restrict__ s0,
                             const float* __restrict__ s1,
                             const float* __restrict__ s2,
                             const float* __restrict__ s3,
                             const float* __restrict__ s4,
                             const float* __restrict__ s5,
                             u16* __restrict__ dst) {
  const int mat = blockIdx.y;
  const float* src = s0;
  switch (mat) {
    case 1: src = s1; break; case 2: src = s2; break;
    case 3: src = s3; break; case 4: src = s4; break;
    case 5: src = s5; break; default: break;
  }
  const size_t p = (size_t)blockIdx.x * TPB + threadIdx.x;  // 0..MATSZ-1
  const int nt = (int)(p >> 14);
  const int r  = (int)(p & 16383);
  const int kt = r >> 9;
  const int q  = r & 511;
  const int L  = q >> 4;
  const int i  = q & 15;
  const int n  = nt * 16 + (L & 15);
  const int k  = kt * 32 + ((i >> 3) << 4) + ((L >> 4) << 3) + (i & 7);
  dst[(size_t)mat * MATSZ + p] = f2bf(src[(size_t)n * HDIM + k]);
}

// ---------------------------------------------------------------------------
// Persistent seq2seq kernel
// ---------------------------------------------------------------------------
__global__ __launch_bounds__(TPB, 1)
void lstm_persist(const float* __restrict__ prev, const float* __restrict__ nfut,
                  const unsigned char* __restrict__ tf,
                  const float* __restrict__ encWih0,
                  const float* __restrict__ eb0i, const float* __restrict__ eb0h,
                  const float* __restrict__ eb1i, const float* __restrict__ eb1h,
                  const float* __restrict__ decWih0,
                  const float* __restrict__ db0i, const float* __restrict__ db0h,
                  const float* __restrict__ db1i, const float* __restrict__ db1h,
                  const float* __restrict__ fcw, const float* __restrict__ fcb,
                  const u16* __restrict__ pEWhh0, const u16* __restrict__ pEWih1,
                  const u16* __restrict__ pEWhh1, const u16* __restrict__ pDWhh0,
                  const u16* __restrict__ pDWih1, const u16* __restrict__ pDWhh1,
                  u16* __restrict__ h0g, u16* __restrict__ h1g,
                  float* __restrict__ xbuf, u32* __restrict__ cnt,
                  float* __restrict__ out) {
  __shared__ u16   hs[32 * HP];            // 66 KB staged h (bf16)
  __shared__ float gates[4 * 32 * GPCH];   // 18 KB f32 gate tiles
  __shared__ float c0s[BSZ * 32];          // WG-private cell states
  __shared__ float c1s[BSZ * 32];
  __shared__ float xs[BSZ];
  __shared__ float red[BSZ * 8];
  __shared__ float fcs[HDIM];

  const int tid   = threadIdx.x;
  const int lane  = tid & 31;
  const int wave  = tid >> 5;
  const int gate  = wave >> 1;
  const int nh    = (wave & 1) << 4;
  const int u0    = blockIdx.x * 32;
  const int mbase = (lane >> 4) << 3;
  const int u_l   = nh + (lane & 15);

  for (int i = tid; i < BSZ * 32; i += TPB) { c0s[i] = 0.f; c1s[i] = 0.f; }
  if (blockIdx.x == 0)
    for (int i = tid; i < HDIM; i += TPB) fcs[i] = fcw[i];

  // Per-(wave,lane) constants for its gate column
  const int   gcol = gate * HDIM + u0 + u_l;
  const float eB0 = eb0i[gcol] + eb0h[gcol];
  const float eW0 = encWih0[gcol];
  const float eB1 = eb1i[gcol] + eb1h[gcol];
  const float dB0 = db0i[gcol] + db0h[gcol];
  const float dW0 = decWih0[gcol];
  const float dB1 = db1i[gcol] + db1h[gcol];
  const int   nt  = (gate * HDIM + u0 + nh) >> 4;

  u32 bar = 0;
  int p = 0;                               // h ping-pong parity
  __syncthreads();

  // ============================= ENCODER ==================================
  for (int t = 0; t < TLEN; ++t) {
    // ---- layer 0: gates = x_t*Wih0 + h0@Whh0^T + b ----
    if (tid < BSZ) xs[tid] = prev[tid * TLEN + t];
    stage_h(hs, h0g + p * (BSZ * HDIM), tid);
    __syncthreads();
    v8f acc0, acc1;
#pragma unroll
    for (int r = 0; r < 8; ++r) {
      acc0[r] = eB0 + xs[mbase + r] * eW0;
      acc1[r] = eB0 + xs[16 + mbase + r] * eW0;
    }
    gemm_pass(pEWhh0, hs, lane, nt, acc0, acc1);
    store_gates(gates, acc0, acc1, gate, u_l, mbase);
    __syncthreads();
    cell_update(gates, c0s, h0g + (1 - p) * (BSZ * HDIM), u0, tid);
    grid_sync(cnt, (++bar) * NWG);

    // ---- layer 1: gates = h0_new@Wih1^T + h1@Whh1^T + b ----
    stage_h(hs, h0g + (1 - p) * (BSZ * HDIM), tid);
    __syncthreads();
#pragma unroll
    for (int r = 0; r < 8; ++r) { acc0[r] = eB1; acc1[r] = eB1; }
    gemm_pass(pEWih1, hs, lane, nt, acc0, acc1);
    __syncthreads();
    stage_h(hs, h1g + p * (BSZ * HDIM), tid);
    __syncthreads();
    gemm_pass(pEWhh1, hs, lane, nt, acc0, acc1);
    store_gates(gates, acc0, acc1, gate, u_l, mbase);
    __syncthreads();
    cell_update(gates, c1s, h1g + (1 - p) * (BSZ * HDIM), u0, tid);
    // no grid sync needed: next step's L0 sync orders all h1 writes before
    // any WG can enter the next L1 phase (bounded-skew argument)
    p ^= 1;
  }

  // ============================= DECODER ==================================
  for (int t = 0; t < TLEN; ++t) {
    // ---- layer 0 (x from feedback buffer; x_0 = 0 from init kernel) ----
    if (tid < BSZ) xs[tid] = xbuf[tid];
    stage_h(hs, h0g + p * (BSZ * HDIM), tid);
    __syncthreads();
    v8f acc0, acc1;
#pragma unroll
    for (int r = 0; r < 8; ++r) {
      acc0[r] = dB0 + xs[mbase + r] * dW0;
      acc1[r] = dB0 + xs[16 + mbase + r] * dW0;
    }
    gemm_pass(pDWhh0, hs, lane, nt, acc0, acc1);
    store_gates(gates, acc0, acc1, gate, u_l, mbase);
    __syncthreads();
    cell_update(gates, c0s, h0g + (1 - p) * (BSZ * HDIM), u0, tid);
    grid_sync(cnt, (++bar) * NWG);

    // ---- layer 1 ----
    stage_h(hs, h0g + (1 - p) * (BSZ * HDIM), tid);
    __syncthreads();
#pragma unroll
    for (int r = 0; r < 8; ++r) { acc0[r] = dB1; acc1[r] = dB1; }
    gemm_pass(pDWih1, hs, lane, nt, acc0, acc1);
    __syncthreads();
    stage_h(hs, h1g + p * (BSZ * HDIM), tid);
    __syncthreads();
    gemm_pass(pDWhh1, hs, lane, nt, acc0, acc1);
    store_gates(gates, acc0, acc1, gate, u_l, mbase);
    __syncthreads();
    cell_update(gates, c1s, h1g + (1 - p) * (BSZ * HDIM), u0, tid);
    grid_sync(cnt, (++bar) * NWG);

    // ---- fc head + teacher forcing (WG0 only) ----
    if (blockIdx.x == 0) {
      const int m = tid >> 3, part = tid & 7;
      const u16* h1n = h1g + (1 - p) * (BSZ * HDIM) + m * HDIM;
      float s = 0.f;
      for (int u = part; u < HDIM; u += 8) s += bf2f(h1n[u]) * fcs[u];
      red[m * 8 + part] = s;
      __syncthreads();
      if (part == 0) {
        float pr = fcb[0];
#pragma unroll
        for (int k = 0; k < 8; ++k) pr += red[m * 8 + k];
        out[m * TLEN + t] = pr;
        xbuf[m] = tf[t] ? nfut[m * TLEN + t] : pr;
      }
    }
    grid_sync(cnt, (++bar) * NWG);
    p ^= 1;
  }
}

// ---------------------------------------------------------------------------
// Workspace layout (bytes):
//   [0,256)            grid-sync counter
//   [256,512)          x feedback buffer (32 f32)
//   [512, +128KB)      h0 ping-pong (2 x 32x1024 bf16)
//   [+128KB, +256KB)   h1 ping-pong
//   [512KB, 512KB+48MB) 6 packed bf16 weight matrices (8MB each)
// Requires ws_size >= ~51 MB.
// ---------------------------------------------------------------------------
extern "C" void kernel_launch(void* const* d_in, const int* in_sizes, int n_in,
                              void* d_out, int out_size, void* d_ws,
                              size_t ws_size, hipStream_t stream) {
  (void)in_sizes; (void)n_in; (void)out_size; (void)ws_size;
  const float* prev    = (const float*)d_in[0];
  const float* nfut    = (const float*)d_in[1];
  const unsigned char* tf = (const unsigned char*)d_in[2];
  const float* encWih0 = (const float*)d_in[3];
  const float* encWhh0 = (const float*)d_in[4];
  const float* eb0i    = (const float*)d_in[5];
  const float* eb0h    = (const float*)d_in[6];
  const float* encWih1 = (const float*)d_in[7];
  const float* encWhh1 = (const float*)d_in[8];
  const float* eb1i    = (const float*)d_in[9];
  const float* eb1h    = (const float*)d_in[10];
  const float* decWih0 = (const float*)d_in[11];
  const float* decWhh0 = (const float*)d_in[12];
  const float* db0i    = (const float*)d_in[13];
  const float* db0h    = (const float*)d_in[14];
  const float* decWih1 = (const float*)d_in[15];
  const float* decWhh1 = (const float*)d_in[16];
  const float* db1i    = (const float*)d_in[17];
  const float* db1h    = (const float*)d_in[18];
  const float* fcw     = (const float*)d_in[19];
  const float* fcb     = (const float*)d_in[20];

  char* ws = (char*)d_ws;
  u32* cnt  = (u32*)(ws + 0);
  float* xbuf = (float*)(ws + 256);
  u16* h0g  = (u16*)(ws + 512);
  u16* h1g  = (u16*)(ws + 512 + 131072);
  u16* pk   = (u16*)(ws + (1u << 19));
  u16* pEWhh0 = pk + 0 * (size_t)MATSZ;
  u16* pEWih1 = pk + 1 * (size_t)MATSZ;
  u16* pEWhh1 = pk + 2 * (size_t)MATSZ;
  u16* pDWhh0 = pk + 3 * (size_t)MATSZ;
  u16* pDWih1 = pk + 4 * (size_t)MATSZ;
  u16* pDWhh1 = pk + 5 * (size_t)MATSZ;

  // 1) zero counter + x feedback + h ping-pong buffers (262656 B = 65664 u32)
  zero_u32_kernel<<<257, TPB, 0, stream>>>((u32*)ws, 65664);
  // 2) convert + pack weights into WMMA fragment order
  pack6_kernel<<<dim3(MATSZ / TPB, 6), TPB, 0, stream>>>(
      encWhh0, encWih1, encWhh1, decWhh0, decWih1, decWhh1, pk);
  // 3) persistent seq2seq
  lstm_persist<<<NWG, TPB, 0, stream>>>(
      prev, nfut, tf, encWih0, eb0i, eb0h, eb1i, eb1h, decWih0, db0i, db0h,
      db1i, db1h, fcw, fcb, pEWhh0, pEWih1, pEWhh1, pDWhh0, pDWih1, pDWhh1,
      h0g, h1g, xbuf, cnt, (float*)d_out);
}
